// HGraphSAGE_78786880078302
// MI455X (gfx1250) — matrix operand order repaired and checked by reading, not compiled, not run on hardware
//
#include <hip/hip_runtime.h>

#define D 128
#define TILE_ROWS 64

typedef __attribute__((ext_vector_type(16))) __bf16 v16bf;
typedef __attribute__((ext_vector_type(8)))  __bf16 v8bf;
typedef __attribute__((ext_vector_type(8)))  float  v8f;

__device__ __forceinline__ unsigned short f2bf(float f) {
    // round-to-nearest-even float32 -> bf16
    unsigned int u = __float_as_uint(f);
    u += 0x7FFFu + ((u >> 16) & 1u);
    return (unsigned short)(u >> 16);
}

__global__ void zero_f32(float* __restrict__ p, int n) {
    int i = blockIdx.x * blockDim.x + threadIdx.x;
    int stride = gridDim.x * blockDim.x;
    for (; i < n; i += stride) p[i] = 0.0f;
}

// wt_self[e*D+k] = bf16(sum_r Ws[r,k,e])   (transposed: column-major for B-frag loads)
// wt_n[r][e*D+k] = bf16(Wn[r,k,e])
// bsum[e] = sum_r b[r,e]
__global__ void prep_weights(const float* __restrict__ Ws, const float* __restrict__ Wn,
                             const float* __restrict__ b,
                             unsigned short* __restrict__ wt_self,
                             unsigned short* __restrict__ wt_n,
                             float* __restrict__ bsum, int R_) {
    int idx = blockIdx.x * blockDim.x + threadIdx.x;
    if (idx >= D * D) return;
    int k = idx / D, e = idx % D;
    float s = 0.f;
    for (int r = 0; r < R_; ++r) s += Ws[(size_t)r * D * D + k * D + e];
    wt_self[e * D + k] = f2bf(s);
    for (int r = 0; r < R_; ++r)
        wt_n[(size_t)r * D * D + e * D + k] = f2bf(Wn[(size_t)r * D * D + k * D + e]);
    if (idx < D) {
        float bs = 0.f;
        for (int r = 0; r < R_; ++r) bs += b[r * D + idx];
        bsum[idx] = bs;
    }
}

// 32 lanes per edge, 4 floats per lane: agg[dst] += x[src]; lane 0 also counts degree.
__global__ void scatter_kernel(const float* __restrict__ x, const int* __restrict__ src,
                               const int* __restrict__ dst, float* __restrict__ agg,
                               float* __restrict__ deg, int E_) {
    long long tid = (long long)blockIdx.x * blockDim.x + threadIdx.x;
    if (tid >= (long long)E_ * 32) return;
    int e = (int)(tid >> 5);
    int q = (int)(tid & 31);
    int s = src[e];
    int dd = dst[e];
    float4 v = ((const float4*)(x + (size_t)s * D))[q];
    float* o = agg + (size_t)dd * D + q * 4;
    __hip_atomic_fetch_add(o + 0, v.x, __ATOMIC_RELAXED, __HIP_MEMORY_SCOPE_AGENT);
    __hip_atomic_fetch_add(o + 1, v.y, __ATOMIC_RELAXED, __HIP_MEMORY_SCOPE_AGENT);
    __hip_atomic_fetch_add(o + 2, v.z, __ATOMIC_RELAXED, __HIP_MEMORY_SCOPE_AGENT);
    __hip_atomic_fetch_add(o + 3, v.w, __ATOMIC_RELAXED, __HIP_MEMORY_SCOPE_AGENT);
    if (q == 0)
        __hip_atomic_fetch_add(deg + dd, 1.0f, __ATOMIC_RELAXED, __HIP_MEMORY_SCOPE_AGENT);
}

// out[N x 128] (op)= (A * rowscale) @ W, W given transposed bf16 as Wt[col][k].
// MODE 0: out = acc + bias[col]    (initial write, self GEMM)
// MODE 1: out += acc               (accumulate neighbor GEMM)
// MODE 2: out = relu(out + acc)    (last neighbor GEMM of layer 1)
template <int MODE>
__global__ void gemm_wmma(const float* __restrict__ A, const unsigned short* __restrict__ Wt,
                          const float* __restrict__ rowscale, const float* __restrict__ bias,
                          float* __restrict__ out, int nrows) {
    __shared__ unsigned short As[TILE_ROWS * D];  // 16 KB, A tile in bf16 (mean-scaled)
    int row0 = blockIdx.x * TILE_ROWS;
    int t = threadIdx.x;

    // Stage A tile: 64 rows x 128 cols, fp32 -> bf16, with optional 1/max(deg,1) row scale.
#pragma unroll
    for (int i = 0; i < 8; ++i) {
        int idx = t + 256 * i;        // 0..2047
        int lr  = idx >> 5;           // local row 0..63
        int c4  = idx & 31;           // float4 chunk within row
        int gr  = row0 + lr;
        float4 v = make_float4(0.f, 0.f, 0.f, 0.f);
        float sc = 1.0f;
        if (gr < nrows) {
            v = *(const float4*)(A + (size_t)gr * D + c4 * 4);
            if (rowscale) sc = 1.0f / fmaxf(rowscale[gr], 1.0f);
        }
        unsigned short* p = &As[lr * D + c4 * 4];
        p[0] = f2bf(v.x * sc);
        p[1] = f2bf(v.y * sc);
        p[2] = f2bf(v.z * sc);
        p[3] = f2bf(v.w * sc);
    }
    __syncthreads();

    int wave = t >> 5, lane = t & 31;
    int half = lane >> 4, r = lane & 15;
    int n0 = wave * 16;  // this wave's 16-column slice
    int gc = n0 + r;
    const bool full = (row0 + TILE_ROWS) <= nrows;
    float bv = (MODE == 0) ? bias[gc] : 0.0f;

    // B fragments: lane holds column n0+r; half 0 -> K[k0..k0+15], half 1 -> K[k0+16..k0+31]
    v16bf bfrag[4];
    const unsigned short* wcol = Wt + (size_t)gc * D;
#pragma unroll
    for (int kc = 0; kc < 4; ++kc)
        bfrag[kc] = *(const v16bf*)(wcol + kc * 32 + half * 16);

#pragma unroll
    for (int m = 0; m < 4; ++m) {
        v8f c = {};
#pragma unroll
        for (int kc = 0; kc < 4; ++kc) {
            // A frag: lane = row 16*m+r; half h holds K = k0+16h..+15 (contiguous)
            const unsigned short* ap = &As[(16 * m + r) * D + kc * 32 + half * 16];
            v8bf lo = *(const v8bf*)(ap);
            v8bf hi = *(const v8bf*)(ap + 8);
            v16bf a;
#pragma unroll
            for (int j = 0; j < 8; ++j) { a[j] = lo[j]; a[j + 8] = hi[j]; }
            c = __builtin_amdgcn_wmma_f32_16x16x32_bf16(
                    false, a, false, bfrag[kc], (short)0, c, false, false);
        }
        // Epilogue: C VGPR i -> row (8*half + i), col = gc
        float* obase = out + (size_t)(row0 + 16 * m + half * 8) * D + gc;
        if (full) {
#pragma unroll
            for (int i = 0; i < 8; ++i) {
                float* o = obase + (size_t)i * D;
                float v = c[i];
                if (MODE == 0)      *o = v + bv;
                else if (MODE == 1) *o = *o + v;
                else                *o = fmaxf(*o + v, 0.0f);
            }
        } else {
#pragma unroll
            for (int i = 0; i < 8; ++i) {
                int gr = row0 + 16 * m + half * 8 + i;
                if (gr < nrows) {
                    float* o = obase + (size_t)i * D;
                    float v = c[i];
                    if (MODE == 0)      *o = v + bv;
                    else if (MODE == 1) *o = *o + v;
                    else                *o = fmaxf(*o + v, 0.0f);
                }
            }
        }
    }
}

extern "C" void kernel_launch(void* const* d_in, const int* in_sizes, int n_in,
                              void* d_out, int out_size, void* d_ws, size_t ws_size,
                              hipStream_t stream) {
    const float* x   = (const float*)d_in[0];
    const int*   src = (const int*)d_in[1];
    const int*   dst = (const int*)d_in[2];
    const float* Ws1 = (const float*)d_in[3];
    const float* Wn1 = (const float*)d_in[4];
    const float* b1  = (const float*)d_in[5];
    const float* Ws2 = (const float*)d_in[6];
    const float* Wn2 = (const float*)d_in[7];
    const float* b2  = (const float*)d_in[8];

    const int N = in_sizes[0] / D;           // 100000
    const int R = in_sizes[3] / (D * D);     // 3
    const int E = in_sizes[1] / R;           // 800000

    char* ws = (char*)d_ws;
    float* agg  = (float*)ws;                                   // N*D f32
    float* h1   = (float*)(ws + (size_t)N * D * 4);             // N*D f32
    float* deg  = (float*)(ws + (size_t)N * D * 8);             // N f32
    float* bsum = (float*)(ws + (size_t)N * D * 8 + (size_t)N * 4);
    unsigned short* wt_self = (unsigned short*)((char*)bsum + 512);
    unsigned short* wt_n    = wt_self + D * D;

    float* out = (float*)d_out;
    const int gemm_blocks = (N + TILE_ROWS - 1) / TILE_ROWS;
    const int scat_blocks = (int)(((long long)E * 32 + 255) / 256);

    auto run_layer = [&](const float* xin, float* xout, const float* Ws,
                         const float* Wn, const float* b, bool relu) {
        prep_weights<<<(D * D + 255) / 256, 256, 0, stream>>>(Ws, Wn, b, wt_self, wt_n, bsum, R);
        // self path: xout = xin @ sum_r(Ws_r) + sum_r(b_r)
        gemm_wmma<0><<<gemm_blocks, 256, 0, stream>>>(xin, wt_self, nullptr, bsum, xout, N);
        for (int r = 0; r < R; ++r) {
            zero_f32<<<4096, 256, 0, stream>>>(agg, N * D);
            zero_f32<<<256, 256, 0, stream>>>(deg, N);
            scatter_kernel<<<scat_blocks, 256, 0, stream>>>(xin, src + (size_t)r * E,
                                                            dst + (size_t)r * E, agg, deg, E);
            bool last = (r == R - 1);
            if (last && relu)
                gemm_wmma<2><<<gemm_blocks, 256, 0, stream>>>(agg, wt_n + (size_t)r * D * D,
                                                              deg, nullptr, xout, N);
            else
                gemm_wmma<1><<<gemm_blocks, 256, 0, stream>>>(agg, wt_n + (size_t)r * D * D,
                                                              deg, nullptr, xout, N);
        }
    };

    run_layer(x,  h1,  Ws1, Wn1, b1, true);
    run_layer(h1, out, Ws2, Wn2, b2, false);
}
